// Network_9929964388466
// MI455X (gfx1250) — matrix operand order repaired and checked by reading, not compiled, not run on hardware
//
#include <hip/hip_runtime.h>
#include <hip/hip_bf16.h>

// ---------------- problem constants (from reference) ----------------
#define B_DIM   16
#define C_DIM   16384
#define T_DIM   300
#define O1_DIM  240
#define O2_DIM  10
#define NTOT    (B_DIM * T_DIM)   // 4800 columns (b,t)
#define KLEN    64
#define THETA   10.0f
#define KCHUNK  64                // K staging chunk for gemm1
#define XROW    (KCHUNK + 8)      // LDS row stride (bf16) -> 144B
#define S1_STRIDE 240             // s1 stored [n][o], o contiguous (480B rows)
#define W2_STRIDE 256             // W2 bf16 padded to 16x256

typedef __attribute__((ext_vector_type(16))) __bf16 v16bf;
typedef __attribute__((ext_vector_type(8)))  __bf16 v8bf;
typedef __attribute__((ext_vector_type(4)))  __bf16 v4bf;
typedef __attribute__((ext_vector_type(2)))  __bf16 v2bf;
typedef __attribute__((ext_vector_type(8)))  float  v8f;
typedef __attribute__((ext_vector_type(4)))  unsigned int v4u;
typedef __attribute__((ext_vector_type(8)))  int    v8i;
typedef __attribute__((ext_vector_type(4)))  int    v4i;

union V16U { v16bf v; struct { v8bf lo, hi; } p; };

// =====================================================================
// Weight conversions to bf16 in workspace (tiny; every call, deterministic).
// =====================================================================
__global__ __launch_bounds__(256) void cvt_w1_kernel(
    const float* __restrict__ W1, __bf16* __restrict__ W1b) {
  const size_t i = ((size_t)blockIdx.x * 256 + threadIdx.x) * 4;
  const float4 f = *(const float4*)(W1 + i);
  v4bf o = { (__bf16)f.x, (__bf16)f.y, (__bf16)f.z, (__bf16)f.w };
  *(v4bf*)(W1b + i) = o;
}

__global__ __launch_bounds__(256) void cvt_w2_kernel(
    const float* __restrict__ W2, __bf16* __restrict__ W2b) {
  const int i = blockIdx.x * 256 + threadIdx.x;    // 16*256 = 4096
  const int m = i >> 8, c = i & 255;
  W2b[i] = (m < O2_DIM && c < O1_DIM) ? (__bf16)W2[m * O1_DIM + c] : (__bf16)0.0f;
}

// =====================================================================
// GEMM1: u1pre[o][n] = sum_c W1[o][c] * x[b][c][t],  n = b*300+t
// 300 N-blocks of 16 cols; 256 threads = 8 waves; 2 M-tiles per wave.
// Double-buffered LDS staging of x (fp32 -> packed bf16), [n][c] layout so
// B fragments are 2x ds_load_b128; A fragments 2x global_load_b128 from
// L2-resident bf16 W1. One barrier per K-chunk.
// =====================================================================
__global__ __launch_bounds__(256) void gemm1_kernel(
    const float* __restrict__ x, const __bf16* __restrict__ W1b,
    float* __restrict__ u1pre) {
  __shared__ __bf16 xs[2][16][XROW];           // [buf][n_local][c_local]

  const int tid  = threadIdx.x;
  const int wave = tid >> 5;
  const int lane = tid & 31;
  const int n0   = blockIdx.x * 16;

  const int lrow  = lane & 15;
  const int lhalf = lane >> 4;
  const int kbA   = lhalf * 8;                 // A: K groups kbA.., kbA+16..
  const int kbB   = lhalf * 16;                // unused; B loads use kbA, kbA+16 like A

  const int mt0 = wave * 2;
  int mA0 = mt0 * 16 + lrow;       if (mA0 >= O1_DIM) mA0 = O1_DIM - 1;
  int mA1 = mt0 * 16 + 16 + lrow;  if (mA1 >= O1_DIM) mA1 = O1_DIM - 1;
  const __bf16* __restrict__ a0p = W1b + (size_t)mA0 * C_DIM;
  const __bf16* __restrict__ a1p = W1b + (size_t)mA1 * C_DIM;

  // cooperative x loader: tid -> (c-pair = (tid>>4)*2, n = tid&15)
  const int ln  = tid & 15;
  const int lc2 = (tid >> 4) * 2;
  const int gn = n0 + ln;
  const int gb = gn / T_DIM;
  const int gt = gn - gb * T_DIM;
  const size_t xcol = (size_t)gb * C_DIM * T_DIM + (size_t)gt;

  auto stage = [&](int buf, int k0) {
#pragma unroll
    for (int i = 0; i < 2; ++i) {
      const int cl = lc2 + i * 32;
      const float f0 = x[xcol + (size_t)(k0 + cl) * T_DIM];
      const float f1 = x[xcol + (size_t)(k0 + cl + 1) * T_DIM];
      v2bf pk = { (__bf16)f0, (__bf16)f1 };
      *(v2bf*)&xs[buf][ln][cl] = pk;
    }
  };

  v8f acc0 = {}; v8f acc1 = {};

  stage(0, 0);
  int cur = 0;
  for (int k0 = 0; k0 < C_DIM; k0 += KCHUNK) {
    __syncthreads();                           // staging(cur) done; reads(prev) done
    if (k0 + KCHUNK < C_DIM) {
      __builtin_prefetch(&x[xcol + (size_t)(k0 + 2 * KCHUNK + lc2) * T_DIM], 0, 3);
      stage(cur ^ 1, k0 + KCHUNK);
    }
#pragma unroll
    for (int ks = 0; ks < KCHUNK / 32; ++ks) {
      const int kk = k0 + ks * 32;
      V16U bfr;
      bfr.p.lo = *(const v8bf*)&xs[cur][lrow][ks * 32 + kbA];
      bfr.p.hi = *(const v8bf*)&xs[cur][lrow][ks * 32 + kbA + 16];
      V16U a0, a1;
      a0.p.lo = *(const v8bf*)(a0p + kk + kbA);
      a0.p.hi = *(const v8bf*)(a0p + kk + kbA + 16);
      a1.p.lo = *(const v8bf*)(a1p + kk + kbA);
      a1.p.hi = *(const v8bf*)(a1p + kk + kbA + 16);
      acc0 = __builtin_amdgcn_wmma_f32_16x16x32_bf16(false, a0.v, false, bfr.v,
                                                     (short)0, acc0, false, false);
      acc1 = __builtin_amdgcn_wmma_f32_16x16x32_bf16(false, a1.v, false, bfr.v,
                                                     (short)0, acc1, false, false);
    }
    cur ^= 1;
  }

  const int ncol = n0 + lrow;
#pragma unroll
  for (int r = 0; r < 8; ++r) {
    const int m0 = mt0 * 16 + r + lhalf * 8;
    const int m1 = (mt0 + 1) * 16 + r + lhalf * 8;
    if (m0 < O1_DIM) u1pre[(size_t)m0 * NTOT + ncol] = acc0[r];
    if (m1 < O1_DIM) u1pre[(size_t)m1 * NTOT + ncol] = acc1[r];
  }
}

// =====================================================================
// psp (64-tap alpha FIR) + sequential spike scan with alpha refractory.
// One thread per neuron (b,o); LDS circular buffers hist/fut [idx][tid].
// =====================================================================
template <typename OT>
__global__ __launch_bounds__(64) void spike_kernel(
    const float* __restrict__ upre, OT* __restrict__ out,
    int nNeur, int O, long ob, long oo, long ot) {
  __shared__ float srm[KLEN];
  __shared__ float refk[KLEN];
  __shared__ float hist[KLEN][64];
  __shared__ float fut[KLEN][64];

  const int tid = threadIdx.x;
  {  // eps(t) = (t/10)e^{1-t/10};  nu(t) = -20 t e^{1-t}
    const float tt = (float)tid;
    srm[tid]  = (tt * 0.1f) * __expf(1.0f - tt * 0.1f);
    refk[tid] = -2.0f * THETA * tt * __expf(1.0f - tt);
  }
#pragma unroll
  for (int i = 0; i < KLEN; ++i) fut[i][tid] = 0.0f;
  __syncthreads();

  const int  nid    = blockIdx.x * 64 + tid;
  const bool active = nid < nNeur;
  const int  b      = active ? nid / O : 0;
  const int  o      = active ? nid - b * O : 0;
  const float* __restrict__ up = upre + (size_t)o * NTOT + (size_t)b * T_DIM;
  OT* __restrict__ op = out + (long)b * ob + (long)o * oo;

  for (int t = 0; t < T_DIM; ++t) {
    hist[t & 63][tid] = active ? up[t] : 0.0f;
    float u = 0.0f;
    const int kmax = (t < KLEN - 1) ? t : (KLEN - 1);
    for (int k = 0; k <= kmax; ++k) u += srm[k] * hist[(t - k) & 63][tid];
    const float v = u + fut[t & 63][tid];
    fut[t & 63][tid] = 0.0f;
    const float s = (v >= THETA) ? 1.0f : 0.0f;
    if (s != 0.0f) {
      for (int k = 1; k < KLEN; ++k) fut[(t + k) & 63][tid] += refk[k];
    }
    if (active) op[(long)t * ot] = (OT)s;
  }
}

// =====================================================================
// GEMM2: u2pre[o2][n] = sum_o W2[o2][o] * s1[n][o]; M=16(10 valid),
// K=256 (240 valid), N=4800. One wave per 16-column tile. The 16x240
// bf16 B tile is DMA'd into LDS by the Tensor Data Mover (2D D#:
// tile_dim0=240 contiguous o, tile_dim1=16 n-rows, stride=240, 2B elems),
// completion via s_wait_tensorcnt, then 8 WMMA K-steps from LDS.
// =====================================================================
__global__ __launch_bounds__(32) void gemm2_kernel(
    const __bf16* __restrict__ s1, const __bf16* __restrict__ W2b,
    float* __restrict__ u2pre) {
  __shared__ __bf16 bs[16][S1_STRIDE];         // 7680 B, rows 480B (16B-aligned)

  const int lane = threadIdx.x;
  const int n0   = blockIdx.x * 16;

  // ---- Tensor DMA descriptor (D#) ----
  const unsigned lds_addr = (unsigned)(size_t)&bs[0][0];
  const unsigned long long ga =
      (unsigned long long)(size_t)(s1 + (size_t)n0 * S1_STRIDE);
  v4u g0 = { 1u,                                  // count=1, user descriptor
             lds_addr,                            // lds_addr [63:32]
             (unsigned)(ga & 0xFFFFFFFFu),        // global_addr [95:64]
             (unsigned)((ga >> 32) & 0x01FFFFFFu) | (2u << 30) };  // addr hi | type=2
  v8i g1 = { (int)(1u << 16),                     // wg_mask=0, data_size=1 (2B)
             (int)((S1_STRIDE & 0xFFFF) << 16),   // tensor_dim0 lo16
             (int)(((unsigned)NTOT & 0xFFFFu) << 16),  // dim0 hi16=0 | tensor_dim1 lo16
             (int)((unsigned)S1_STRIDE << 16),    // dim1 hi16=0 | tile_dim0=240
             16,                                  // tile_dim1=16, tile_dim2=0
             S1_STRIDE,                           // tensor_dim0_stride lo32 = 240
             0, 0 };                              // stride0 hi, dim1_stride (unused 2D)
  v4i gz = { 0, 0, 0, 0 };
#if __clang_major__ >= 23
  v8i gz8 = { 0, 0, 0, 0, 0, 0, 0, 0 };
  __builtin_amdgcn_tensor_load_to_lds(g0, g1, gz, gz, gz8, 0);
#else
  __builtin_amdgcn_tensor_load_to_lds(g0, g1, gz, gz, 0);
#endif
  __builtin_amdgcn_s_wait_tensorcnt((short)0);

  const int lrow  = lane & 15;
  const int lhalf = lane >> 4;
  const int kbA   = lhalf * 8;
  const int kbB   = lhalf * 16;
  const int ncol  = n0 + lrow;
  const __bf16* __restrict__ ap = W2b + lrow * W2_STRIDE;
  const v8bf zero = {};

  v8f acc = {};
#pragma unroll
  for (int kk = 0; kk < 256; kk += 32) {
    V16U af, bfr;
    af.p.lo = *(const v8bf*)(ap + kk + kbA);
    af.p.hi = *(const v8bf*)(ap + kk + kbA + 16);
    const int b0 = kk + kbA, b1 = b0 + 16;
    bfr.p.lo = (b0 < O1_DIM) ? *(const v8bf*)&bs[lrow][b0] : zero;
    bfr.p.hi = (b1 < O1_DIM) ? *(const v8bf*)&bs[lrow][b1] : zero;
    acc = __builtin_amdgcn_wmma_f32_16x16x32_bf16(false, af.v, false, bfr.v,
                                                  (short)0, acc, false, false);
  }
#pragma unroll
  for (int r = 0; r < 8; ++r) {
    const int mm = r + lhalf * 8;
    if (mm < O2_DIM) u2pre[(size_t)mm * NTOT + ncol] = acc[r];
  }
}

// =====================================================================
extern "C" void kernel_launch(void* const* d_in, const int* in_sizes, int n_in,
                              void* d_out, int out_size, void* d_ws, size_t ws_size,
                              hipStream_t stream) {
  const float* x  = (const float*)d_in[0];   // (16, 16384, 300)
  const float* W1 = (const float*)d_in[1];   // (240, 16384)
  const float* W2 = (const float*)d_in[2];   // (10, 240)
  float* out = (float*)d_out;                // (16, 10, 300)

  char* ws = (char*)d_ws;
  const size_t U1_BYTES  = (size_t)O1_DIM * NTOT * sizeof(float);       // 4,608,000
  const size_t S1_BYTES  = (size_t)NTOT * S1_STRIDE * sizeof(__bf16);   // 2,304,000
  const size_t U2_BYTES  = (size_t)O2_DIM * NTOT * sizeof(float);       //   192,000
  const size_t W1B_BYTES = (size_t)O1_DIM * C_DIM * sizeof(__bf16);     // 7,864,320
  float*  u1pre = (float*)ws;
  __bf16* s1    = (__bf16*)(ws + U1_BYTES);
  float*  u2pre = (float*)(ws + U1_BYTES + S1_BYTES);
  __bf16* W1b   = (__bf16*)(ws + U1_BYTES + S1_BYTES + U2_BYTES);
  __bf16* W2b   = (__bf16*)(ws + U1_BYTES + S1_BYTES + U2_BYTES + W1B_BYTES);

  cvt_w1_kernel<<<(O1_DIM * C_DIM / 4 + 255) / 256, 256, 0, stream>>>(W1, W1b);
  cvt_w2_kernel<<<16, 256, 0, stream>>>(W2, W2b);

  // layer 1: u1pre = W1 @ x  (HBM-bound: x streamed once, W1b from L2)
  gemm1_kernel<<<NTOT / 16, 256, 0, stream>>>(x, W1b, u1pre);
  // layer 1: s1[n][o] = spike(psp(u1pre)) as bf16
  spike_kernel<__bf16><<<(B_DIM * O1_DIM + 63) / 64, 64, 0, stream>>>(
      u1pre, s1, B_DIM * O1_DIM, O1_DIM,
      /*ob*/ (long)T_DIM * S1_STRIDE, /*oo*/ 1L, /*ot*/ (long)S1_STRIDE);
  // layer 2: u2pre = W2 @ s1 (TDM-staged B tiles)
  gemm2_kernel<<<NTOT / 16, 32, 0, stream>>>(s1, W2b, u2pre);
  // layer 2: out = spike(psp(u2pre)) -> (b, o2, t) row-major
  spike_kernel<float><<<(B_DIM * O2_DIM + 63) / 64, 64, 0, stream>>>(
      u2pre, out, B_DIM * O2_DIM, O2_DIM,
      /*ob*/ (long)O2_DIM * T_DIM, /*oo*/ (long)T_DIM, /*ot*/ 1L);
}
